// MultiHeadAttention_69432441307175
// MI455X (gfx1250) — matrix operand order repaired and checked by reading, not compiled, not run on hardware
//
#include <hip/hip_runtime.h>
#include <hip/hip_bf16.h>

// ---------------------------------------------------------------------------
// MultiHeadAttention (H=1, B=32, G=1024, D=256) for MI455X / gfx1250.
// bf16 WMMA (v_wmma_f32_16x16x32_bf16) everywhere, fp32 accumulate.
// Pipeline: [prepack bf16] -> [QKV proj] -> [flash attention, LDS-staged K/V
// via global_load_async_to_lds_b128] -> [output proj + bias].
// All GEMM orientations chosen so every store is a contiguous 16B/32B write.
// ---------------------------------------------------------------------------

typedef __bf16 bf16_t;
typedef __attribute__((ext_vector_type(16))) __bf16 v16bf;
typedef __attribute__((ext_vector_type(8)))  __bf16 v8bf;
typedef __attribute__((ext_vector_type(4)))  __bf16 v4bf;
typedef __attribute__((ext_vector_type(8)))  float  v8f;
typedef __attribute__((ext_vector_type(4)))  int    v4i;

#define BATCH 32
#define G     1024
#define D     256
#define NT    (D / 16)   // 16 column tiles
#define KF    (D / 32)   // 8 k-fragments of 32

// ---- gfx1250 async global->LDS copy path -----------------------------------
#if defined(__has_builtin)
#  if __has_builtin(__builtin_amdgcn_global_load_async_to_lds_b128) && \
      __has_builtin(__builtin_amdgcn_s_wait_asynccnt)
#    define USE_ASYNC_LDS 1
#  endif
#endif
#ifndef USE_ASYNC_LDS
#  define USE_ASYNC_LDS 0
#endif

#if USE_ASYNC_LDS
typedef __attribute__((address_space(1))) v4i* gv4i_p;   // global v4i*
typedef __attribute__((address_space(3))) v4i* lv4i_p;   // LDS v4i*
#endif

static __device__ __forceinline__ void cp16_g2lds(const bf16_t* gsrc,
                                                  bf16_t* ldst) {
#if USE_ASYNC_LDS
  __builtin_amdgcn_global_load_async_to_lds_b128(
      (gv4i_p)gsrc, (lv4i_p)ldst, /*offset=*/0, /*cpol=*/0);
#else
  *(v8bf*)ldst = *(const v8bf*)gsrc;   // global_load_b128 + ds_store_b128
#endif
}
static __device__ __forceinline__ void cp_wait() {
#if USE_ASYNC_LDS
  __builtin_amdgcn_s_wait_asynccnt(0);
#endif
}

static __device__ __forceinline__ v8f wmma_bf16(v16bf a, v16bf b, v8f c) {
  return __builtin_amdgcn_wmma_f32_16x16x32_bf16(false, a, false, b, (short)0, c,
                                                 false, false);
}

// A fragment (16x32 bf16): lane row = lane&15, k-offset (lane<16?0:16);
// elements [0..7] at p, [8..15] at p+8 (16 contiguous k).
static __device__ __forceinline__ v16bf afrag(const bf16_t* __restrict__ p) {
  v8bf lo = *(const v8bf*)p;
  v8bf up = *(const v8bf*)(p + 8);
  v16bf f;
#pragma unroll
  for (int i = 0; i < 8; ++i) { f[i] = lo[i]; f[8 + i] = up[i]; }
  return f;
}
// B fragment (32x16 bf16, column-major source): lane col = lane&15,
// 16 contiguous elements at k-offset (lane<16?0:16).
static __device__ __forceinline__ v16bf bfrag(const bf16_t* __restrict__ p) {
  return *(const v16bf*)p;
}

// ---------------------------------------------------------------------------
// Stage 0: fp32 -> bf16 prepack (data and the four weight matrices).
// ---------------------------------------------------------------------------
__global__ void __launch_bounds__(256)
cvt_bf16_kernel(const float* __restrict__ src, bf16_t* __restrict__ dst, int n4) {
  const int i = blockIdx.x * blockDim.x + threadIdx.x;
  if (i >= n4) return;
  const float4 f = ((const float4*)src)[i];
  v4bf o;
  o[0] = (bf16_t)f.x; o[1] = (bf16_t)f.y; o[2] = (bf16_t)f.z; o[3] = (bf16_t)f.w;
  ((v4bf*)dst)[i] = o;
}

// ---------------------------------------------------------------------------
// Stage 1: Q = NORM*X@Wq^T, K = X@Wk^T (computed as transposed tiles so the
// store axis is contiguous), V^T = (X@Wv^T)^T (original orientation already
// stores contiguously into Vt). One wave per (proj, 16-token tile).
// ---------------------------------------------------------------------------
__global__ void __launch_bounds__(128)
qkv_proj_kernel(const bf16_t* __restrict__ X,
                const bf16_t* __restrict__ Wq, const bf16_t* __restrict__ Wk,
                const bf16_t* __restrict__ Wv,
                bf16_t* __restrict__ Qbf, bf16_t* __restrict__ Kbf,
                bf16_t* __restrict__ Vt) {
  const int lane = threadIdx.x & 31;
  const int job  = blockIdx.x * 4 + (threadIdx.x >> 5);
  const int proj = job >> 11;          // 0=Q, 1=K, 2=V
  const int mt   = job & 2047;
  const int l15  = lane & 15;
  const int k0a  = (lane < 16) ? 0 : 16;
  const int k0b  = (lane < 16) ? 0 : 16;
  const int ohi  = (lane < 16) ? 0 : 8;

  if (proj < 2) {
    // C^T tiles: A = W rows (out x in), B = X tokens as columns.
    const bf16_t* W = proj ? Wk : Wq;
    bf16_t* dst = proj ? Kbf : Qbf;
    const float scale = proj ? 1.0f : 0.0625f;      // NORM folded into Q
    const int tok = mt * 16 + l15;
    v16bf bx[KF];
#pragma unroll
    for (int kf = 0; kf < KF; ++kf)
      bx[kf] = bfrag(X + (size_t)tok * D + kf * 32 + k0b);
    for (int nt = 0; nt < NT; ++nt) {
      v8f c = {};
#pragma unroll
      for (int kf = 0; kf < KF; ++kf)
        c = wmma_bf16(afrag(W + (size_t)(nt * 16 + l15) * D + kf * 32 + k0a),
                      bx[kf], c);
      v8bf s;
#pragma unroll
      for (int v = 0; v < 8; ++v) s[v] = (bf16_t)(c[v] * scale);
      *(v8bf*)(dst + (size_t)tok * D + nt * 16 + ohi) = s;   // contiguous 16B
    }
  } else {
    // V: A = X rows (tokens), B = Wv rows as columns; store into V^T (dv x G).
    const int row = mt * 16 + l15;
    v16bf ax[KF];
#pragma unroll
    for (int kf = 0; kf < KF; ++kf)
      ax[kf] = afrag(X + (size_t)row * D + kf * 32 + k0a);
    const int b  = (mt * 16) >> 10;
    const int g0 = (mt * 16) & (G - 1);
    for (int nt = 0; nt < NT; ++nt) {
      const int col = nt * 16 + l15;
      v8f c = {};
#pragma unroll
      for (int kf = 0; kf < KF; ++kf)
        c = wmma_bf16(ax[kf], bfrag(Wv + (size_t)col * D + kf * 32 + k0b), c);
      v8bf s;
#pragma unroll
      for (int v = 0; v < 8; ++v) s[v] = (bf16_t)c[v];
      *(v8bf*)(Vt + (size_t)b * D * G + (size_t)col * G + g0 + ohi) = s;
    }
  }
}

// ---------------------------------------------------------------------------
// Stage 2: flash attention. Block = 4 waves, all in the same batch; each wave
// owns one 16-query tile. Per 32-key step the block stages K (32x256) and
// V^T (256x32) into padded LDS (async global->LDS), then:
//   S^T = K.Q^T  (lane = query, keys in VGPRs -> softmax needs 1 shfl_xor(16))
//   O^T += V^T.P^T (P^T as B operand via a 4-dword half-wave exchange;
//                   alpha/1-over-l are scalar per lane).
// Reference mask semantics: fill stays in the denominator, numerator re-zeroed.
// ---------------------------------------------------------------------------
#define KS_LD 264   // 256 + 8 pad (bf16 elems)
#define VS_LD 40    // 32 + 8 pad

__global__ void __launch_bounds__(128)
flash_attn_kernel(const bf16_t* __restrict__ Qbf, const bf16_t* __restrict__ Kbf,
                  const bf16_t* __restrict__ Vt,  const int* __restrict__ mask,
                  const int* __restrict__ evaluate, bf16_t* __restrict__ Hbf) {
  __shared__ __align__(16) bf16_t Ks[32 * KS_LD];    // ~16.9 KB
  __shared__ __align__(16) bf16_t Vs[256 * VS_LD];   // ~20.0 KB

  const int tid  = threadIdx.x;
  const int lane = tid & 31;
  const int wave = tid >> 5;
  const int b    = blockIdx.x >> 4;                  // batch
  const int qt   = (blockIdx.x & 15) * 4 + wave;     // query tile 0..63
  const int l15  = lane & 15;
  const int hi   = lane >> 4;                        // 0 | 1
  const int ohi  = hi ? 8 : 0;
  const int k0a  = hi ? 16 : 0;
  const float fill = (*evaluate) ? -__builtin_inff() : -30.0f;

  const bf16_t* Kb = Kbf + (size_t)b * G * D;
  const bf16_t* Vb = Vt  + (size_t)b * D * G;
  const int*    mb = mask + b * G;

  // Q (pre-scaled by NORM) as B fragments: column = query = lane&15.
  const int qtok = b * G + qt * 16 + l15;
  v16bf qf[KF];
#pragma unroll
  for (int kf = 0; kf < KF; ++kf)
    qf[kf] = bfrag(Qbf + (size_t)qtok * D + kf * 32 + (hi ? 16 : 0));

  v8f o[NT];
#pragma unroll
  for (int nt = 0; nt < NT; ++nt) o[nt] = (v8f){};

  float m_run = -__builtin_inff();
  float l_run = 0.0f;

  const int kr = tid >> 2;   // K-stage: row 0..31
  const int kq = tid & 3;    //          quarter of the 256-wide row

  for (int step = 0; step < G / 32; ++step) {
    const int kbase = step * 32;

    // ---- stage K tile (32 x 256) and V^T tile (256 x 32) into LDS
#pragma unroll
    for (int c = 0; c < 4; ++c) {
      const int off = (kq * 4 + c) * 16;
      cp16_g2lds(Kb + (size_t)(kbase + kr) * D + off, Ks + kr * KS_LD + off);
    }
#pragma unroll
    for (int r = 0; r < 2; ++r) {
      const int row = tid * 2 + r;
      cp16_g2lds(Vb + (size_t)row * G + kbase,      Vs + row * VS_LD);
      cp16_g2lds(Vb + (size_t)row * G + kbase + 16, Vs + row * VS_LD + 16);
    }
    cp_wait();
    __syncthreads();

    // ---- S^T tiles (keys x queries) from LDS
    v8f c0 = {}, c1 = {};
#pragma unroll
    for (int kf = 0; kf < KF; ++kf)
      c0 = wmma_bf16(afrag(Ks + l15 * KS_LD + kf * 32 + k0a), qf[kf], c0);
#pragma unroll
    for (int kf = 0; kf < KF; ++kf)
      c1 = wmma_bf16(afrag(Ks + (16 + l15) * KS_LD + kf * 32 + k0a), qf[kf], c1);

    // ---- mask fill (lane holds keys v+8*hi [tile0] / +16 [tile1], query l15)
    int mk0[8], mk1[8];
#pragma unroll
    for (int v = 0; v < 8; ++v) {
      mk0[v] = mb[kbase + ohi + v];
      mk1[v] = mb[kbase + 16 + ohi + v];
      if (mk0[v]) c0[v] = fill;
      if (mk1[v]) c1[v] = fill;
    }

    // ---- online softmax stats for query l15 (split across lane & lane+16)
    float tmax = fmaxf(c0[0], c1[0]);
#pragma unroll
    for (int v = 1; v < 8; ++v) tmax = fmaxf(tmax, fmaxf(c0[v], c1[v]));
    tmax = fmaxf(tmax, __shfl_xor(tmax, 16, 32));
    const float m_new = fmaxf(m_run, tmax);
    const float alpha = __expf(m_run - m_new);

    float p0[8], p1[8], psum = 0.0f;
#pragma unroll
    for (int v = 0; v < 8; ++v) {
      p0[v] = __expf(c0[v] - m_new);
      p1[v] = __expf(c1[v] - m_new);
      psum += p0[v] + p1[v];                 // fill terms stay in denominator
    }
    psum += __shfl_xor(psum, 16, 32);
    l_run = l_run * alpha + psum;
    m_run = m_new;

    // ---- build P^T B-fragment (col = query, 16 contiguous keys per lane):
    // lane<16 needs {own p0, partner p0}; lane>=16 needs {partner p1, own p1}.
    union PK { v8bf v; int i[4]; };
    PK m0p, m1p, send, recv;
#pragma unroll
    for (int v = 0; v < 8; ++v) {
      m0p.v[v] = (bf16_t)(mk0[v] ? 0.0f : p0[v]);  // numerator re-zeroed
      m1p.v[v] = (bf16_t)(mk1[v] ? 0.0f : p1[v]);
    }
    send = hi ? m0p : m1p;
#pragma unroll
    for (int j = 0; j < 4; ++j) recv.i[j] = __shfl_xor(send.i[j], 16, 32);
    v16bf pf;
#pragma unroll
    for (int v = 0; v < 8; ++v) {
      pf[v]     = hi ? recv.v[v] : m0p.v[v];
      pf[8 + v] = hi ? m1p.v[v]  : recv.v[v];
    }

    // ---- rescale O^T (all elements belong to query l15 -> scalar alpha)
#pragma unroll
    for (int nt = 0; nt < NT; ++nt)
#pragma unroll
      for (int v = 0; v < 8; ++v) o[nt][v] *= alpha;

    // ---- O^T += V^T . P^T   (A = V^T rows from LDS)
#pragma unroll
    for (int nt = 0; nt < NT; ++nt)
      o[nt] = wmma_bf16(afrag(Vs + (nt * 16 + l15) * VS_LD + k0a), pf, o[nt]);

    __syncthreads();   // before next step overwrites the LDS tiles
  }

  // ---- normalize (scalar per lane) and store heads: contiguous 16B writes
  const float inv = 1.0f / l_run;
#pragma unroll
  for (int nt = 0; nt < NT; ++nt) {
    v8bf s;
#pragma unroll
    for (int v = 0; v < 8; ++v) s[v] = (bf16_t)(o[nt][v] * inv);
    *(v8bf*)(Hbf + (size_t)(b * G + qt * 16 + l15) * D + nt * 16 + ohi) = s;
  }
}

// ---------------------------------------------------------------------------
// Stage 3: out = heads @ Wout^T + bout (fp32), transposed tiles so each lane
// stores 8 contiguous floats.
// ---------------------------------------------------------------------------
__global__ void __launch_bounds__(128)
out_proj_kernel(const bf16_t* __restrict__ H, const bf16_t* __restrict__ Wo,
                const float* __restrict__ bo, float* __restrict__ out) {
  const int lane = threadIdx.x & 31;
  const int mt   = blockIdx.x * 4 + (threadIdx.x >> 5);
  const int l15  = lane & 15;
  const int k0a  = (lane < 16) ? 0 : 16;
  const int k0b  = (lane < 16) ? 0 : 16;
  const int ohi  = (lane < 16) ? 0 : 8;
  const int tok  = mt * 16 + l15;

  v16bf bh[KF];
#pragma unroll
  for (int kf = 0; kf < KF; ++kf)
    bh[kf] = bfrag(H + (size_t)tok * D + kf * 32 + k0b);

  for (int nt = 0; nt < NT; ++nt) {
    v8f c = {};
#pragma unroll
    for (int kf = 0; kf < KF; ++kf)
      c = wmma_bf16(afrag(Wo + (size_t)(nt * 16 + l15) * D + kf * 32 + k0a),
                    bh[kf], c);
    v8f r;
#pragma unroll
    for (int v = 0; v < 8; ++v) r[v] = c[v] + bo[nt * 16 + ohi + v];
    *(v8f*)(out + (size_t)tok * D + nt * 16 + ohi) = r;   // contiguous 32B
  }
}

// ---------------------------------------------------------------------------
extern "C" void kernel_launch(void* const* d_in, const int* in_sizes, int n_in,
                              void* d_out, int out_size, void* d_ws, size_t ws_size,
                              hipStream_t stream) {
  const float* data     = (const float*)d_in[0];
  const int*   mask     = (const int*)  d_in[1];
  const int*   evaluate = (const int*)  d_in[3];
  const float* Wq       = (const float*)d_in[4];
  const float* Wk       = (const float*)d_in[5];
  const float* Wv       = (const float*)d_in[6];
  const float* Wo       = (const float*)d_in[7];
  const float* bo       = (const float*)d_in[8];

  const size_t TOKD = (size_t)BATCH * G * D;   // 8388608 elems
  bf16_t* Qbf = (bf16_t*)d_ws;
  bf16_t* Kbf = Qbf + TOKD;
  bf16_t* Vt  = Kbf + TOKD;
  bf16_t* Hbf = Vt  + TOKD;
  bf16_t* Xbf = Hbf + TOKD;
  bf16_t* Wqb = Xbf + TOKD;
  bf16_t* Wkb = Wqb + D * D;
  bf16_t* Wvb = Wkb + D * D;
  bf16_t* Wob = Wvb + D * D;
  // workspace total: 5*16 MiB + 4*128 KiB = 80.5 MiB

  // Stage 0: prepack to bf16
  cvt_bf16_kernel<<<(int)(TOKD / 4 / 256), 256, 0, stream>>>(data, Xbf, (int)(TOKD / 4));
  cvt_bf16_kernel<<<64, 256, 0, stream>>>(Wq, Wqb, D * D / 4);
  cvt_bf16_kernel<<<64, 256, 0, stream>>>(Wk, Wkb, D * D / 4);
  cvt_bf16_kernel<<<64, 256, 0, stream>>>(Wv, Wvb, D * D / 4);
  cvt_bf16_kernel<<<64, 256, 0, stream>>>(Wo, Wob, D * D / 4);

  // Stage 1: 3 projections x 2048 token tiles, 1 wave each
  qkv_proj_kernel<<<1536, 128, 0, stream>>>(Xbf, Wqb, Wkb, Wvb, Qbf, Kbf, Vt);
  // Stage 2: 32 batches x 64 query tiles (4 waves/block share a batch)
  flash_attn_kernel<<<512, 128, 0, stream>>>(Qbf, Kbf, Vt, mask, evaluate, Hbf);
  // Stage 3: 2048 token tiles
  out_proj_kernel<<<512, 128, 0, stream>>>(Hbf, Wob, bo, (float*)d_out);
}
